// TransformerTranslation_3075196583960
// MI455X (gfx1250) — matrix operand order repaired and checked by reading, not compiled, not run on hardware
//
#include <hip/hip_runtime.h>

// ---------------------------------------------------------------------------
// Tiny encoder-decoder transformer forward on MI455X (gfx1250, wave32).
// GEMMs:  v_wmma_f32_16x16x32_f16 (f16 in, f32 accumulate).
// Weights staged Global->LDS with CDNA5 async-LDS DMA (ASYNCcnt) via inline asm.
// ---------------------------------------------------------------------------

#define D_EMB   64
#define N_HEAD  4
#define N_LAYER 6
#define VOCAB   32000
#define SEQ     256
#define BATCH   16
#define M_TOK   (BATCH * SEQ)   // 4096 tokens

typedef __attribute__((ext_vector_type(16))) _Float16 v16h;
typedef __attribute__((ext_vector_type(8)))  float    v8f;

__device__ __forceinline__ v8f wmma_f16(v16h a, v16h b, v8f c) {
  // (neg_a, A, neg_b, B, c_mod, C, reuse_a, reuse_b)
  return __builtin_amdgcn_wmma_f32_16x16x32_f16(false, a, false, b, (short)0, c,
                                                false, false);
}

// Async Global->LDS 16-byte copy (CDNA5 GLOBAL_LOAD_ASYNC_TO_LDS_B128).
// ldsoff: byte offset within the wave's LDS allocation (low 32 bits of a
// generic pointer to __shared__); gaddr: 16B-aligned global address.
__device__ __forceinline__ void async_g2l_b128(unsigned ldsoff, const float* gaddr) {
  asm volatile("global_load_async_to_lds_b128 %0, %1, off"
               :: "v"(ldsoff), "v"(gaddr) : "memory");
}
__device__ __forceinline__ void wait_async0() {
  asm volatile("s_wait_asynccnt 0" ::: "memory");
}
__device__ __forceinline__ unsigned lds_off(const void* p) {
  return (unsigned)(unsigned long long)p;   // generic ptr low 32 bits = LDS offset
}

// ---------------------------------------------------------------------------
// Embedding: enc = tok_emb_in[idx] + pos ; dec = tok_emb_out[shift(targets)] + pos
// (decoder deliberately reuses pos_emb_in, matching the reference)
// ---------------------------------------------------------------------------
__global__ void embed_kernel(const int* __restrict__ idx, const int* __restrict__ tgt,
                             const float* __restrict__ tokin,
                             const float* __restrict__ tokout,
                             const float* __restrict__ pos,
                             float* __restrict__ encx, float* __restrict__ decx) {
  int i = blockIdx.x * blockDim.x + threadIdx.x;
  if (i >= M_TOK * D_EMB) return;
  int d  = i & (D_EMB - 1);
  int bt = i >> 6;                 // token index in [0, B*T)
  int t  = bt & (SEQ - 1);
  int te = idx[bt];
  int td = (t == 0) ? 0 : tgt[bt - 1];   // <SOS>=0 then targets[:, :-1]
  float p = pos[t * D_EMB + d];
  encx[i] = tokin[(size_t)te * D_EMB + d] + p;
  decx[i] = tokout[(size_t)td * D_EMB + d] + p;
}

// ---------------------------------------------------------------------------
// LayerNorm over D=64: one wave32 per token, 2 elements per lane.
// ---------------------------------------------------------------------------
__global__ void ln_kernel(const float* __restrict__ X, float* __restrict__ Y,
                          const float* __restrict__ g, const float* __restrict__ bta) {
  int token = blockIdx.x * (blockDim.x >> 5) + (threadIdx.x >> 5);
  int lane  = threadIdx.x & 31;
  const float* x = X + (size_t)token * D_EMB;
  float a0 = x[lane], a1 = x[lane + 32];
  float s = a0 + a1;
#pragma unroll
  for (int off = 16; off > 0; off >>= 1) s += __shfl_xor(s, off, 32);
  float mean = s * (1.f / D_EMB);
  float d0 = a0 - mean, d1 = a1 - mean;
  float vv = d0 * d0 + d1 * d1;
#pragma unroll
  for (int off = 16; off > 0; off >>= 1) vv += __shfl_xor(vv, off, 32);
  float rstd = rsqrtf(vv * (1.f / D_EMB) + 1e-5f);
  Y[(size_t)token * D_EMB + lane]      = d0 * rstd * g[lane]      + bta[lane];
  Y[(size_t)token * D_EMB + lane + 32] = d1 * rstd * g[lane + 32] + bta[lane + 32];
}

// ---------------------------------------------------------------------------
// Generic token GEMM: C[M_TOK,N] = act(A[M_TOK,K] @ W[K,N] + bias) + residual
// W staged Global->LDS (f32) with async-LDS DMA. One 16x16 D-tile per wave,
// K consumed 32 at a time by v_wmma_f32_16x16x32_f16.
// A-frag (16-bit A 16x32): lane l -> row m=l%16; e<8: k=8*(l/16)+e,
//                          e>=8: k=16+8*(l/16)+(e-8).
// B-frag: lane l -> col n=l%16; e<8: k=8*(l/16)+e, e>=8: k=16+8*(l/16)+(e-8).
// C/D: vgpr r, lane l -> m = r + 8*(l/16), n = l%16.
// ---------------------------------------------------------------------------
template <bool HAS_BIAS, bool RELU, bool HAS_RES>
__global__ void gemm_kernel(const float* __restrict__ A, const float* __restrict__ W,
                            const float* __restrict__ bias,
                            const float* __restrict__ residual,
                            float* __restrict__ C, int N, int K) {
  extern __shared__ float sW[];                 // K*N f32
  int tid = threadIdx.x;
  int total = K * N;                            // multiple of blockDim.x*4
  for (int i = tid * 4; i < total; i += blockDim.x * 4)
    async_g2l_b128(lds_off(sW + i), W + i);
  wait_async0();
  __syncthreads();

  int wave = tid >> 5, lane = tid & 31;
  int half = lane >> 4, l16 = lane & 15;
  int tilesN = N >> 4;
  int tile = blockIdx.x * (blockDim.x >> 5) + wave;   // grids sized exactly
  int tm = (tile / tilesN) << 4;
  int tn = (tile % tilesN) << 4;

  v8f acc = {};
  const float* arow = A + (size_t)(tm + l16) * K;
  for (int kk = 0; kk < K; kk += 32) {
    float4 alo = *(const float4*)(arow + kk + half * 8);
    float4 al1 = *(const float4*)(arow + kk + half * 8 + 4);
    float4 ahi = *(const float4*)(arow + kk + 16 + half * 8);
    float4 ah1 = *(const float4*)(arow + kk + 16 + half * 8 + 4);
    v16h a, b;
    a[0] = (_Float16)alo.x; a[1] = (_Float16)alo.y; a[2] = (_Float16)alo.z; a[3] = (_Float16)alo.w;
    a[4] = (_Float16)al1.x; a[5] = (_Float16)al1.y; a[6] = (_Float16)al1.z; a[7] = (_Float16)al1.w;
    a[8]  = (_Float16)ahi.x; a[9]  = (_Float16)ahi.y; a[10] = (_Float16)ahi.z; a[11] = (_Float16)ahi.w;
    a[12] = (_Float16)ah1.x; a[13] = (_Float16)ah1.y; a[14] = (_Float16)ah1.z; a[15] = (_Float16)ah1.w;
#pragma unroll
    for (int e = 0; e < 16; ++e)
      b[e] = (_Float16)sW[(kk + ((e & 8) << 1) + half * 8 + (e & 7)) * N + tn + l16];
    acc = wmma_f16(a, b, acc);
  }

  int n = tn + l16;
  float bn = HAS_BIAS ? bias[n] : 0.f;
#pragma unroll
  for (int vr = 0; vr < 8; ++vr) {
    int m = tm + vr + 8 * half;
    float v = acc[vr] + bn;
    if (RELU) v = fmaxf(v, 0.f);
    if (HAS_RES) v += residual[(size_t)m * N + n];
    C[(size_t)m * N + n] = v;
  }
}

// ---------------------------------------------------------------------------
// Attention for one (batch, head, 16-query tile): scores = Q K^T * D^-0.5,
// optional causal mask, exact softmax, then P @ V. head_dim=16 is zero-padded
// to K=32 for the QK^T WMMA. One wave per block.
// All global fragment loads are unconditional float4 (b128) loads; lane-half
// zeroing is done with selects AFTER the load so no predicated VMEM chains.
// ---------------------------------------------------------------------------
__global__ void attn_kernel(const float* __restrict__ Q, const float* __restrict__ Kb,
                            const float* __restrict__ Vb, float* __restrict__ O,
                            int causal) {
  __shared__ float    sS[16][SEQ];   // raw scores (fp32)
  __shared__ _Float16 sP[16][SEQ];   // softmax probabilities (f16)
  int blk = blockIdx.x;
  int mt = blk & 15;                 // query tile within sequence (T/16 = 16)
  int h  = (blk >> 4) & (N_HEAD - 1);
  int b  = blk >> 6;
  int lane = threadIdx.x & 31;
  int half = lane >> 4, l16 = lane & 15;
  size_t base = ((size_t)b * SEQ) * D_EMB + h * 16;
  const _Float16 hz = (_Float16)0.f;

  // A-frag of Q: real d = 0..15, upper half of K=32 zeroed.
  v16h qf;
  {
    const float* qrow = Q + base + (size_t)(mt * 16 + l16) * D_EMB + half * 8;
    float4 q0 = *(const float4*)(qrow);
    float4 q1 = *(const float4*)(qrow + 4);
    qf[0] = (_Float16)q0.x; qf[1] = (_Float16)q0.y; qf[2] = (_Float16)q0.z; qf[3] = (_Float16)q0.w;
    qf[4] = (_Float16)q1.x; qf[5] = (_Float16)q1.y; qf[6] = (_Float16)q1.z; qf[7] = (_Float16)q1.w;
#pragma unroll
    for (int e = 8; e < 16; ++e) qf[e] = hz;
  }

  // scores over all 16 key tiles
  bool lo = (half == 0);
  for (int j = 0; j < 16; ++j) {
    const float* krow = Kb + base + (size_t)(j * 16 + l16) * D_EMB;
    float4 k0 = *(const float4*)(krow);        // valid for every lane
    float4 k1 = *(const float4*)(krow + 4);
    float4 k2 = *(const float4*)(krow + 8);
    float4 k3 = *(const float4*)(krow + 12);
    v16h kf;                                   // B-frag of K^T: k-dim = d
    kf[0]  = lo ? (_Float16)k0.x : (_Float16)k2.x; kf[1]  = lo ? (_Float16)k0.y : (_Float16)k2.y;
    kf[2]  = lo ? (_Float16)k0.z : (_Float16)k2.z; kf[3]  = lo ? (_Float16)k0.w : (_Float16)k2.w;
    kf[4]  = lo ? (_Float16)k1.x : (_Float16)k3.x; kf[5]  = lo ? (_Float16)k1.y : (_Float16)k3.y;
    kf[6]  = lo ? (_Float16)k1.z : (_Float16)k3.z; kf[7]  = lo ? (_Float16)k1.w : (_Float16)k3.w;
    kf[8]  = hz; kf[9]  = hz;
    kf[10] = hz; kf[11] = hz;
    kf[12] = hz; kf[13] = hz;
    kf[14] = hz; kf[15] = hz;
    v8f c = {};
    c = wmma_f16(qf, kf, c);
#pragma unroll
    for (int vr = 0; vr < 8; ++vr) {
      int ml = vr + 8 * half;
      int ng = j * 16 + l16;
      float s = c[vr] * 0.125f;            // C = n_embd = 64 -> 64^-0.5
      if (causal && ng > mt * 16 + ml) s = -1e30f;
      sS[ml][ng] = s;
    }
  }
  __syncthreads();

  // softmax: lane pair (l16, l16+16) cooperates on row l16
  {
    int m = l16, n0 = half * 128;
    float mx = -3.4e38f;
    for (int nn = 0; nn < 128; ++nn) mx = fmaxf(mx, sS[m][n0 + nn]);
    mx = fmaxf(mx, __shfl_xor(mx, 16, 32));
    float sum = 0.f;
    for (int nn = 0; nn < 128; ++nn) sum += __expf(sS[m][n0 + nn] - mx);
    sum += __shfl_xor(sum, 16, 32);
    float inv = 1.f / sum;
    for (int nn = 0; nn < 128; ++nn)
      sP[m][n0 + nn] = (_Float16)(__expf(sS[m][n0 + nn] - mx) * inv);
  }
  __syncthreads();

  // out = P[16,256] @ V[256,16] : 8 chained WMMAs over K
  v8f acc = {};
  for (int kk = 0; kk < SEQ; kk += 32) {
    v16h a, vf;
#pragma unroll
    for (int e = 0; e < 8; ++e) {
      a[e]     = sP[l16][kk + half * 8 + e];
      a[e + 8] = sP[l16][kk + 16 + half * 8 + e];
    }
#pragma unroll
    for (int e = 0; e < 16; ++e)
      vf[e] = (_Float16)Vb[base + (size_t)(kk + ((e & 8) << 1) + half * 8 + (e & 7)) * D_EMB + l16];
    acc = wmma_f16(a, vf, acc);
  }
#pragma unroll
  for (int vr = 0; vr < 8; ++vr)
    O[base + (size_t)(mt * 16 + vr + 8 * half) * D_EMB + l16] = acc[vr];
}

// ---------------------------------------------------------------------------
// LM head: logits[4096,32000] = X[4096,64] @ lm_w[64,32000] + lm_b.
// Block = 8 waves covering one 16x128 tile; lm_w + X tiles staged into LDS
// with async-LDS DMA (f32), converted to f16 at fragment build.
// ---------------------------------------------------------------------------
__global__ void lmhead_kernel(const float* __restrict__ X, const float* __restrict__ Wl,
                              const float* __restrict__ bl, float* __restrict__ out) {
  __shared__ float sW[D_EMB * 128];    // 32 KB
  __shared__ float sA[16 * D_EMB];     //  4 KB
  int tid = threadIdx.x;
  int mt = blockIdx.x;                 // 0..255  (M tile)
  int n0 = blockIdx.y * 128;           // 0..249 * 128 (V = 250*128)
  for (int i = tid * 4; i < D_EMB * 128; i += blockDim.x * 4) {
    int k = i >> 7, nn = i & 127;      // nn multiple of 4 -> 16B aligned
    async_g2l_b128(lds_off(sW + i), Wl + (size_t)k * VOCAB + n0 + nn);
  }
  for (int i = tid * 4; i < 16 * D_EMB; i += blockDim.x * 4)
    async_g2l_b128(lds_off(sA + i), X + (size_t)mt * 16 * D_EMB + i);
  wait_async0();
  __syncthreads();

  int wave = tid >> 5, lane = tid & 31;
  int half = lane >> 4, l16 = lane & 15;
  int nl = wave * 16 + l16;

  v8f acc = {};
#pragma unroll
  for (int kk = 0; kk < D_EMB; kk += 32) {
    v16h a, b;
#pragma unroll
    for (int e = 0; e < 8; ++e) {
      a[e]     = (_Float16)sA[l16 * D_EMB + kk + half * 8 + e];
      a[e + 8] = (_Float16)sA[l16 * D_EMB + kk + 16 + half * 8 + e];
    }
#pragma unroll
    for (int e = 0; e < 16; ++e)
      b[e] = (_Float16)sW[(kk + ((e & 8) << 1) + half * 8 + (e & 7)) * 128 + nl];
    acc = wmma_f16(a, b, acc);
  }
  float bn = bl[n0 + nl];
#pragma unroll
  for (int vr = 0; vr < 8; ++vr) {
    int m = mt * 16 + vr + 8 * half;
    out[(size_t)m * VOCAB + n0 + nl] = acc[vr] + bn;
  }
}

// ---------------------------------------------------------------------------
// Cross-entropy over the raw logits (log-softmax per 32000-wide row).
// ---------------------------------------------------------------------------
__global__ void zerof_kernel(float* p) { p[0] = 0.f; }

__global__ void loss_kernel(const float* __restrict__ logits,
                            const int* __restrict__ targets,
                            float* __restrict__ accum) {
  __shared__ float red[256];
  int row = blockIdx.x;
  const float* lr = logits + (size_t)row * VOCAB;
  float mx = -3.4e38f;
  for (int i = threadIdx.x; i < VOCAB; i += 256) {
    __builtin_prefetch(&lr[i + 4096], 0, 1);
    mx = fmaxf(mx, lr[i]);
  }
  red[threadIdx.x] = mx;
  __syncthreads();
  for (int s = 128; s > 0; s >>= 1) {
    if (threadIdx.x < s) red[threadIdx.x] = fmaxf(red[threadIdx.x], red[threadIdx.x + s]);
    __syncthreads();
  }
  mx = red[0];
  __syncthreads();
  float sum = 0.f;
  for (int i = threadIdx.x; i < VOCAB; i += 256) sum += __expf(lr[i] - mx);
  red[threadIdx.x] = sum;
  __syncthreads();
  for (int s = 128; s > 0; s >>= 1) {
    if (threadIdx.x < s) red[threadIdx.x] += red[threadIdx.x + s];
    __syncthreads();
  }
  if (threadIdx.x == 0) {
    float lse = mx + __logf(red[0]);
    float lp = lr[targets[row]] - lse;
    atomicAdd(accum, -lp);
  }
}

__global__ void finalize_kernel(const float* __restrict__ accum, float* __restrict__ out) {
  out[0] = accum[0] * (1.f / (float)M_TOK);
}

// ---------------------------------------------------------------------------
// Host orchestration
// ---------------------------------------------------------------------------
extern "C" void kernel_launch(void* const* d_in, const int* in_sizes, int n_in,
                              void* d_out, int out_size, void* d_ws, size_t ws_size,
                              hipStream_t stream) {
  (void)in_sizes; (void)n_in; (void)out_size; (void)ws_size;

  const float* tok_emb_in  = (const float*)d_in[0];
  const float* pos_emb_in  = (const float*)d_in[1];
  const float* tok_emb_out = (const float*)d_in[2];
  const float* enc_wq = (const float*)d_in[3];
  const float* enc_wk = (const float*)d_in[4];
  const float* enc_wv = (const float*)d_in[5];
  const float* enc_pw = (const float*)d_in[6];
  const float* enc_pb = (const float*)d_in[7];
  const float* enc_w1 = (const float*)d_in[8];
  const float* enc_b1 = (const float*)d_in[9];
  const float* enc_w2 = (const float*)d_in[10];
  const float* enc_b2 = (const float*)d_in[11];
  const float* enc_ln1g = (const float*)d_in[12];
  const float* enc_ln1b = (const float*)d_in[13];
  const float* enc_ln2g = (const float*)d_in[14];
  const float* enc_ln2b = (const float*)d_in[15];
  const float* dec_wq = (const float*)d_in[16];
  const float* dec_wk = (const float*)d_in[17];
  const float* dec_wv = (const float*)d_in[18];
  const float* dec_pw = (const float*)d_in[19];
  const float* dec_pb = (const float*)d_in[20];
  const float* ca_wq = (const float*)d_in[21];
  const float* ca_wk = (const float*)d_in[22];
  const float* ca_wv = (const float*)d_in[23];
  const float* ca_pw = (const float*)d_in[24];
  const float* ca_pb = (const float*)d_in[25];
  const float* dec_w1 = (const float*)d_in[26];
  const float* dec_b1 = (const float*)d_in[27];
  const float* dec_w2 = (const float*)d_in[28];
  const float* dec_b2 = (const float*)d_in[29];
  const float* dec_ln1g = (const float*)d_in[30];
  const float* dec_ln1b = (const float*)d_in[31];
  const float* dec_ln2g = (const float*)d_in[32];
  const float* dec_ln2b = (const float*)d_in[33];
  const float* dec_ln3g = (const float*)d_in[34];
  const float* dec_ln3b = (const float*)d_in[35];
  const float* dec_ln4g = (const float*)d_in[36];
  const float* dec_ln4b = (const float*)d_in[37];
  const float* lnf_g = (const float*)d_in[38];
  const float* lnf_b = (const float*)d_in[39];
  const float* lm_w  = (const float*)d_in[40];
  const float* lm_b  = (const float*)d_in[41];
  const int*   idx     = (const int*)d_in[42];
  const int*   targets = (const int*)d_in[43];

  float* logits = (float*)d_out;

  // workspace carve-out (fp32): ~12.6 MB total
  float* ws    = (float*)d_ws;
  float* enc_x = ws;  ws += (size_t)M_TOK * D_EMB;
  float* dec_x = ws;  ws += (size_t)M_TOK * D_EMB;
  float* hbuf  = ws;  ws += (size_t)M_TOK * D_EMB;
  float* h2buf = ws;  ws += (size_t)M_TOK * D_EMB;
  float* qb    = ws;  ws += (size_t)M_TOK * D_EMB;
  float* kb    = ws;  ws += (size_t)M_TOK * D_EMB;
  float* vb    = ws;  ws += (size_t)M_TOK * D_EMB;
  float* ao    = ws;  ws += (size_t)M_TOK * D_EMB;
  float* mid   = ws;  ws += (size_t)M_TOK * 4 * D_EMB;
  float* lossa = ws;  ws += 1;

  auto LN = [&](const float* X, float* Y, const float* g, const float* bta) {
    ln_kernel<<<M_TOK / 8, 256, 0, stream>>>(X, Y, g, bta);
  };
  // variants: plain (QKV), bias+residual (proj / ffn2), bias+relu (ffn1)
  auto GEMM_PLAIN = [&](const float* A, const float* W, float* C, int N, int K) {
    int blocks = (M_TOK / 16) * (N / 16) / 8;
    size_t lds = (size_t)K * N * sizeof(float);
    gemm_kernel<false, false, false><<<blocks, 256, lds, stream>>>(A, W, nullptr, nullptr, C, N, K);
  };
  auto GEMM_BR = [&](const float* A, const float* W, const float* bias,
                     const float* res, float* C, int N, int K) {
    int blocks = (M_TOK / 16) * (N / 16) / 8;
    size_t lds = (size_t)K * N * sizeof(float);
    gemm_kernel<true, false, true><<<blocks, 256, lds, stream>>>(A, W, bias, res, C, N, K);
  };
  auto GEMM_RELU = [&](const float* A, const float* W, const float* bias,
                       float* C, int N, int K) {
    int blocks = (M_TOK / 16) * (N / 16) / 8;
    size_t lds = (size_t)K * N * sizeof(float);
    gemm_kernel<true, true, false><<<blocks, 256, lds, stream>>>(A, W, bias, nullptr, C, N, K);
  };
  auto ATTN = [&](const float* Qp, const float* Kp, const float* Vp, float* O,
                  int causal) {
    attn_kernel<<<BATCH * N_HEAD * (SEQ / 16), 32, 0, stream>>>(Qp, Kp, Vp, O, causal);
  };

  // embeddings
  embed_kernel<<<(M_TOK * D_EMB + 255) / 256, 256, 0, stream>>>(
      idx, targets, tok_emb_in, tok_emb_out, pos_emb_in, enc_x, dec_x);

  const size_t WSZ = (size_t)D_EMB * D_EMB;        // 4096
  const size_t FSZ = (size_t)D_EMB * 4 * D_EMB;    // 16384

  // ---------------- encoder ----------------
  for (int i = 0; i < N_LAYER; ++i) {
    LN(enc_x, hbuf, enc_ln1g + i * D_EMB, enc_ln1b + i * D_EMB);
    GEMM_PLAIN(hbuf, enc_wq + i * WSZ, qb, D_EMB, D_EMB);
    GEMM_PLAIN(hbuf, enc_wk + i * WSZ, kb, D_EMB, D_EMB);
    GEMM_PLAIN(hbuf, enc_wv + i * WSZ, vb, D_EMB, D_EMB);
    ATTN(qb, kb, vb, ao, /*causal=*/0);
    GEMM_BR(ao, enc_pw + i * WSZ, enc_pb + i * D_EMB, enc_x, enc_x, D_EMB, D_EMB);
    LN(enc_x, hbuf, enc_ln2g + i * D_EMB, enc_ln2b + i * D_EMB);
    GEMM_RELU(hbuf, enc_w1 + i * FSZ, enc_b1 + i * 4 * D_EMB, mid, 4 * D_EMB, D_EMB);
    GEMM_BR(mid, enc_w2 + i * FSZ, enc_b2 + i * D_EMB, enc_x, enc_x, D_EMB, 4 * D_EMB);
  }

  // ---------------- decoder ----------------
  for (int i = 0; i < N_LAYER; ++i) {
    // causal self-attention
    LN(dec_x, hbuf, dec_ln1g + i * D_EMB, dec_ln1b + i * D_EMB);
    GEMM_PLAIN(hbuf, dec_wq + i * WSZ, qb, D_EMB, D_EMB);
    GEMM_PLAIN(hbuf, dec_wk + i * WSZ, kb, D_EMB, D_EMB);
    GEMM_PLAIN(hbuf, dec_wv + i * WSZ, vb, D_EMB, D_EMB);
    ATTN(qb, kb, vb, ao, /*causal=*/1);
    GEMM_BR(ao, dec_pw + i * WSZ, dec_pb + i * D_EMB, dec_x, dec_x, D_EMB, D_EMB);
    // cross-attention: queries from decoder, K/V from LN'd encoder output
    LN(enc_x, h2buf, dec_ln2g + i * D_EMB, dec_ln2b + i * D_EMB);
    LN(dec_x, hbuf,  dec_ln3g + i * D_EMB, dec_ln3b + i * D_EMB);
    GEMM_PLAIN(hbuf,  ca_wq + i * WSZ, qb, D_EMB, D_EMB);
    GEMM_PLAIN(h2buf, ca_wk + i * WSZ, kb, D_EMB, D_EMB);
    GEMM_PLAIN(h2buf, ca_wv + i * WSZ, vb, D_EMB, D_EMB);
    ATTN(qb, kb, vb, ao, /*causal=*/0);
    GEMM_BR(ao, ca_pw + i * WSZ, ca_pb + i * D_EMB, dec_x, dec_x, D_EMB, D_EMB);
    // FFN
    LN(dec_x, hbuf, dec_ln4g + i * D_EMB, dec_ln4b + i * D_EMB);
    GEMM_RELU(hbuf, dec_w1 + i * FSZ, dec_b1 + i * 4 * D_EMB, mid, 4 * D_EMB, D_EMB);
    GEMM_BR(mid, dec_w2 + i * FSZ, dec_b2 + i * D_EMB, dec_x, dec_x, D_EMB, 4 * D_EMB);
  }

  // final LN + LM head + loss
  LN(dec_x, hbuf, lnf_g, lnf_b);
  lmhead_kernel<<<dim3(M_TOK / 16, VOCAB / 128), 256, 0, stream>>>(hbuf, lm_w, lm_b, logits);
  zerof_kernel<<<1, 1, 0, stream>>>(lossa);
  loss_kernel<<<M_TOK, 256, 0, stream>>>(logits, targets, lossa);
  finalize_kernel<<<1, 1, 0, stream>>>(lossa, logits + (size_t)M_TOK * VOCAB);
}
